// GIN_2layer_71949292143002
// MI455X (gfx1250) — matrix operand order repaired and checked by reading, not compiled, not run on hardware
//
#include <hip/hip_runtime.h>
#include <hip/hip_bf16.h>

#define NNODES 100000
#define NEDGES 1600000
#define FEAT 64
#define HIDDEN 64
#define CLASSES 32

typedef _Float16 v4h  __attribute__((ext_vector_type(4)));
typedef _Float16 v8h  __attribute__((ext_vector_type(8)));
typedef _Float16 v16h __attribute__((ext_vector_type(16)));
typedef float    v8f  __attribute__((ext_vector_type(8)));

// ---------------- zero fill (agg buffers; d_ws is poisoned) ----------------
__global__ void zero_f32(float* __restrict__ p, int n4) {
    int i = blockIdx.x * blockDim.x + threadIdx.x;
    if (i < n4) reinterpret_cast<float4*>(p)[i] = make_float4(0.f, 0.f, 0.f, 0.f);
}

// ---------------- f32 -> f16 weight conversion ----------------
__global__ void cvt_f32_to_f16(const float* __restrict__ in, _Float16* __restrict__ out, int n) {
    int i = blockIdx.x * blockDim.x + threadIdx.x;
    if (i < n) out[i] = (_Float16)in[i];
}

// ---------------- scatter-add: agg[dst] += feat[src], 16 threads/edge ----------------
__global__ void scatter_add64(const float* __restrict__ feat,
                              const long long* __restrict__ edges,  // [2, NEDGES] int64
                              float* __restrict__ agg) {
    int gid = blockIdx.x * blockDim.x + threadIdx.x;
    int e = gid >> 4;
    if (e >= NEDGES) return;
    int q = (gid & 15) << 2;                  // feature quad offset
    int s = (int)edges[e];                    // src (message source)
    int d = (int)edges[NEDGES + e];           // dst (aggregation target)
    const float4 v = *reinterpret_cast<const float4*>(feat + (size_t)s * FEAT + q);
    float* dp = agg + (size_t)d * FEAT + q;
    atomicAdd(dp + 0, v.x);
    atomicAdd(dp + 1, v.y);
    atomicAdd(dp + 2, v.z);
    atomicAdd(dp + 3, v.w);
}

// ---------------- h16 = f16(a + b), vectorized by 4 ----------------
__global__ void add_cvt_f16(const float* __restrict__ a, const float* __restrict__ b,
                            _Float16* __restrict__ o, int n4) {
    int i = blockIdx.x * blockDim.x + threadIdx.x;
    if (i >= n4) return;
    float4 va = reinterpret_cast<const float4*>(a)[i];
    float4 vb = reinterpret_cast<const float4*>(b)[i];
    v4h r = { (_Float16)(va.x + vb.x), (_Float16)(va.y + vb.y),
              (_Float16)(va.z + vb.z), (_Float16)(va.w + vb.w) };
    reinterpret_cast<v4h*>(o)[i] = r;
}

// ---------------- WMMA GEMM: out[m,n] = relu?( A[m,:] . W[n,:] + bias[n] ) ----------------
// One wave computes a 16-row tile across all NOUT columns. K = 64 (2 WMMA/ntile).
// A: [NNODES,64] f16 row-major; W: [NOUT,64] f16 row-major (torch convention: y = x @ W.T + b)
template <int NOUT, bool RELU>
__global__ void gin_wmma_gemm(const _Float16* __restrict__ A,
                              const _Float16* __restrict__ W,
                              const float* __restrict__ bias,
                              float* __restrict__ out) {
    const int lane = threadIdx.x & 31;
    const int wave = blockIdx.x * (blockDim.x >> 5) + (threadIdx.x >> 5);
    if (wave >= NNODES / 16) return;  // wave-uniform: EXEC stays all-1s for WMMA
    const int m0  = wave << 4;
    const int l15 = lane & 15;
    const bool hi = lane >= 16;

    // --- A fragments (16x32 f16, contiguous K to match B) ---
    // lanes 0-15: VGPR0-7 = K 0..15
    // lanes 16-31: VGPR0-7 = K 16..31   (chunk 1: +32)
    const int aoff = hi ? 16 : 0;
    const _Float16* ap = A + (size_t)(m0 + l15) * 64;
    v8h a0lo = *(const v8h*)(ap + 0  + aoff);
    v8h a0hi = *(const v8h*)(ap + 8  + aoff);
    v8h a1lo = *(const v8h*)(ap + 32 + aoff);
    v8h a1hi = *(const v8h*)(ap + 40 + aoff);
    v16h a0 = __builtin_shufflevector(a0lo, a0hi, 0,1,2,3,4,5,6,7,8,9,10,11,12,13,14,15);
    v16h a1 = __builtin_shufflevector(a1lo, a1hi, 0,1,2,3,4,5,6,7,8,9,10,11,12,13,14,15);

    // --- B fragments: lane n holds W[n, Kbase..Kbase+15] contiguously ---
    const int boff = hi ? 16 : 0;

#pragma unroll
    for (int t = 0; t < NOUT / 16; ++t) {
        const int n = t * 16 + l15;
        const _Float16* wp = W + (size_t)n * 64 + boff;
        v16h b0 = *(const v16h*)(wp);        // K chunk 0 (0..31)
        v16h b1 = *(const v16h*)(wp + 32);   // K chunk 1 (32..63)

        const float bv = bias[n];
        v8f c = { bv, bv, bv, bv, bv, bv, bv, bv };
        c = __builtin_amdgcn_wmma_f32_16x16x32_f16(false, a0, false, b0, (short)0, c, false, false);
        c = __builtin_amdgcn_wmma_f32_16x16x32_f16(false, a1, false, b1, (short)0, c, false, false);

        // D layout: VGPR r -> row m0 + r (+8 for lanes 16-31), col = n
        const int rbase = m0 + (hi ? 8 : 0);
#pragma unroll
        for (int r = 0; r < 8; ++r) {
            float v = c[r];
            if (RELU) v = v > 0.f ? v : 0.f;
            out[(size_t)(rbase + r) * NOUT + n] = v;
        }
    }
}

extern "C" void kernel_launch(void* const* d_in, const int* in_sizes, int n_in,
                              void* d_out, int out_size, void* d_ws, size_t ws_size,
                              hipStream_t stream) {
    const float*     x     = (const float*)d_in[0];
    const long long* edges = (const long long*)d_in[1];   // int64 [2, NEDGES]
    const float*     W1    = (const float*)d_in[2];
    const float*     b1    = (const float*)d_in[3];
    const float*     W2    = (const float*)d_in[4];
    const float*     b2    = (const float*)d_in[5];
    float*           out   = (float*)d_out;               // [NNODES, CLASSES] f32

    // ---- workspace carving (~64 MB) ----
    char* ws = (char*)d_ws;
    const size_t featBytes = (size_t)NNODES * FEAT * sizeof(float);     // 25.6 MB
    float*    agg = (float*)(ws);                                       // reused both layers
    float*    h1  = (float*)(ws + featBytes);                           // layer-1 output, f32
    _Float16* h16 = (_Float16*)(ws + 2 * featBytes);                    // f16 GEMM input, reused
    _Float16* W1h = (_Float16*)(ws + 2 * featBytes + (size_t)NNODES * FEAT * sizeof(_Float16));
    _Float16* W2h = W1h + HIDDEN * FEAT;

    const int n4     = NNODES * FEAT / 4;                 // 1.6M float4s
    const int zb     = (n4 + 255) / 256;
    const int sb     = (NEDGES * 16) / 256;               // 100000 blocks
    const int mtiles = NNODES / 16;                       // 6250 waves
    const int gb     = (mtiles + 7) / 8;                  // 8 waves / 256-thread block

    // weights -> f16 (independent of layer-1 pipeline)
    cvt_f32_to_f16<<<(HIDDEN * FEAT + 255) / 256, 256, 0, stream>>>(W1, W1h, HIDDEN * FEAT);
    cvt_f32_to_f16<<<(CLASSES * HIDDEN + 255) / 256, 256, 0, stream>>>(W2, W2h, CLASSES * HIDDEN);

    // ---- layer 1: h1 = relu((x + seg_sum(x[src])) @ W1.T + b1) ----
    zero_f32<<<zb, 256, 0, stream>>>(agg, n4);
    scatter_add64<<<sb, 256, 0, stream>>>(x, edges, agg);
    add_cvt_f16<<<zb, 256, 0, stream>>>(x, agg, h16, n4);
    gin_wmma_gemm<HIDDEN, true><<<gb, 256, 0, stream>>>(h16, W1h, b1, h1);

    // ---- layer 2: out = (h1 + seg_sum(h1[src])) @ W2.T + b2 ----
    zero_f32<<<zb, 256, 0, stream>>>(agg, n4);
    scatter_add64<<<sb, 256, 0, stream>>>(h1, edges, agg);
    add_cvt_f16<<<zb, 256, 0, stream>>>(h1, agg, h16, n4);
    gin_wmma_gemm<CLASSES, false><<<gb, 256, 0, stream>>>(h16, W2h, b2, out);
}
